// NodeModel_2714419331675
// MI455X (gfx1250) — matrix operand order repaired and checked by reading, not compiled, not run on hardware
//
#include <hip/hip_runtime.h>

typedef __attribute__((ext_vector_type(16))) __bf16        v16bf;
typedef __attribute__((ext_vector_type(2)))  __bf16        v2bf;
typedef __attribute__((ext_vector_type(8)))  float         v8f;
typedef __attribute__((ext_vector_type(8)))  unsigned int  v8u;

// ---------- helpers ----------

__device__ __forceinline__ unsigned short f2bf(float f) {
    unsigned int u = __builtin_bit_cast(unsigned int, f);
    unsigned int r = (u + 0x7FFFu + ((u >> 16) & 1u)) >> 16;   // round-to-nearest-even
    return (unsigned short)r;
}

__device__ __forceinline__ unsigned int pack2(float a, float b) {
    return (unsigned int)f2bf(a) | ((unsigned int)f2bf(b) << 16);
}

__device__ __forceinline__ float4 ld4(const float* p) { return *(const float4*)p; }

__device__ __forceinline__ void st4bf(unsigned short* dst, float4 v) {
    uint2 p; p.x = pack2(v.x, v.y); p.y = pack2(v.z, v.w);
    *(uint2*)dst = p;
}

// 16-byte LDS<-LDS/global register copy (fallback path)
__device__ __forceinline__ void cp16(unsigned short* dst, const unsigned short* src) {
    *(uint4*)dst = *(const uint4*)src;
}

// f32 atomic add, result discarded (relaxed, agent scope -> global_atomic_add_f32)
__device__ __forceinline__ void atomic_add_f32(float* p, float v) {
    (void)__hip_atomic_fetch_add(p, v, __ATOMIC_RELAXED, __HIP_MEMORY_SCOPE_AGENT);
}

// low 32 bits of a generic pointer into __shared__ == wave-relative LDS byte address
__device__ __forceinline__ unsigned lds_addr(const void* p) {
    return (unsigned)(uintptr_t)p;
}

// CDNA5 async copy: LDS[vdst+OFF] = MEM[sbase + voff + OFF], 16 bytes/lane, ASYNCcnt-tracked.
#define ASYNC_CP16(ldst, voff, sbase, OFF)                                      \
    asm volatile("global_load_async_to_lds_b128 %0, %1, %2 offset:" OFF        \
                 :: "v"(ldst), "v"(voff), "s"(sbase) : "memory")

// copy a 128-byte half-row (64 bf16) global -> LDS via 8 async b128 ops
#define ASYNC_CP_HALFROW(ldst, voff, sbase)          \
    do {                                             \
        ASYNC_CP16(ldst, voff, sbase, "0");          \
        ASYNC_CP16(ldst, voff, sbase, "16");         \
        ASYNC_CP16(ldst, voff, sbase, "32");         \
        ASYNC_CP16(ldst, voff, sbase, "48");         \
        ASYNC_CP16(ldst, voff, sbase, "64");         \
        ASYNC_CP16(ldst, voff, sbase, "80");         \
        ASYNC_CP16(ldst, voff, sbase, "96");         \
        ASYNC_CP16(ldst, voff, sbase, "112");        \
    } while (0)

__device__ __forceinline__ v16bf mkfrag(uint4 lo, uint4 hi) {
    v8u v;
    v[0]=lo.x; v[1]=lo.y; v[2]=lo.z; v[3]=lo.w;
    v[4]=hi.x; v[5]=hi.y; v[6]=hi.z; v[7]=hi.w;
    return __builtin_bit_cast(v16bf, v);
}

// A fragment: row-major activation tile; 8 bf16 at klo, 8 bf16 at khi (contiguous k order, matching B).
__device__ __forceinline__ v16bf frag_a(const unsigned short* row, int klo, int khi) {
    return mkfrag(*(const uint4*)(row + klo), *(const uint4*)(row + khi));
}
// B fragment: 16 contiguous bf16 (weights pre-transposed to [N][K]).
__device__ __forceinline__ v16bf frag_b(const unsigned short* p) {
    return mkfrag(*(const uint4*)p, *(const uint4*)(p + 8));
}

__device__ __forceinline__ v8f wmma_bf16(v16bf a, v16bf b, v8f c) {
    return __builtin_amdgcn_wmma_f32_16x16x32_bf16(false, a, false, b, (short)0, c, false, false);
}

#define WAIT_DS()    asm volatile("s_wait_dscnt 0"    ::: "memory")
#define WAIT_ASYNC() asm volatile("s_wait_asynccnt 0" ::: "memory")

// ---------- kernel 0: weights -> bf16 transposed tables ----------

__global__ __launch_bounds__(256) void prep_weights(
    const float* __restrict__ W1a, const float* __restrict__ W1b,
    const float* __restrict__ W2a, const float* __restrict__ W2b,
    unsigned short* __restrict__ wt1a, unsigned short* __restrict__ wt1b,
    unsigned short* __restrict__ wt2a, unsigned short* __restrict__ wt2b)
{
    const int n1 = 160*32, n2 = 32*32, n3 = 224*64, n4 = 64*128;
    int i = blockIdx.x * 256 + threadIdx.x;
    if (i < n1) {
        int k = i / 32, n = i % 32;          wt1a[n*160 + k] = f2bf(W1a[i]);
    } else if (i < n1 + n2) {
        int j = i - n1, k = j / 32, n = j % 32;   wt1b[n*32  + k] = f2bf(W1b[j]);
    } else if (i < n1 + n2 + n3) {
        int j = i - n1 - n2, k = j / 64, n = j % 64;  wt2a[n*224 + k] = f2bf(W2a[j]);
    } else if (i < n1 + n2 + n3 + n4) {
        int j = i - n1 - n2 - n3, k = j / 128, n = j % 128;  wt2b[n*64 + k] = f2bf(W2b[j]);
    }
}

// ---------- kernel 1: f32 -> bf16 streaming cast (for x and u tables) ----------

__global__ __launch_bounds__(256) void prep_cast(
    const float* __restrict__ src, unsigned short* __restrict__ dst, long nquad)
{
    long i = (long)blockIdx.x * 256 + threadIdx.x;
    if (i < nquad) st4bf(dst + i * 4, ld4(src + i * 4));
}

// ---------- kernel 2: zero scatter buffers ----------

__global__ __launch_bounds__(256) void zero_scatter(float* __restrict__ p, int n) {
    int i = blockIdx.x * 256 + threadIdx.x;
    if (i < n) p[i] = 0.0f;
}

// ---------- kernel 3: edge MLP + scatter-add ----------
// One 16-edge tile per wave. 8 waves / block.

#define EDGE_WAVES 8

__global__ __launch_bounds__(256) void edge_kernel(
    const unsigned short* __restrict__ xb,           // [N][128] bf16
    const int* __restrict__ ei, const float* __restrict__ ea,
    const float* __restrict__ b1a, const float* __restrict__ b1b,
    const unsigned short* __restrict__ wt1a, const unsigned short* __restrict__ wt1b,
    float* __restrict__ summed, float* __restrict__ cnt,
    int E, int nTiles)
{
    __shared__ __attribute__((aligned(16))) unsigned short sA[EDGE_WAVES * 16 * 160];
    __shared__ __attribute__((aligned(16))) unsigned short sH[EDGE_WAVES * 16 * 32];
    __shared__ int sCol[EDGE_WAVES * 16];

    const int wv   = threadIdx.x >> 5;
    const int lane = threadIdx.x & 31;
    const int tile = blockIdx.x * EDGE_WAVES + wv;
    if (tile >= nTiles) return;

    const int e    = lane & 15;        // edge-in-tile == WMMA row / col index
    const int half = lane >> 4;        // lane half selects K sub-ranges per ISA layout
    const int eidx = tile * 16 + e;
    const bool valid = (eidx < E);
    const int eldx = valid ? eidx : 0;

    const int r = ei[eldx];            // source node
    const int c = ei[E + eldx];        // destination node
    if (half == 0) {
        sCol[wv * 16 + e] = c;
        if (valid) atomic_add_f32(&cnt[c], 1.0f);
    }

    unsigned short* At = sA + wv * (16 * 160);

    // x part: async bf16 row copy, global -> LDS (no VGPR round-trip, no conversion)
    {
        const unsigned ldst = lds_addr(At + e * 160 + half * 64);
        const unsigned voff = (unsigned)r * 256u + (unsigned)half * 128u;  // bytes into xb
        ASYNC_CP_HALFROW(ldst, voff, xb);
    }
    // edge_attr part: convert f32 -> bf16 in-register
    {
        const float* er = ea + (long)eldx * 32 + half * 16;
        #pragma unroll
        for (int j = 0; j < 4; ++j)
            st4bf(At + e * 160 + 128 + half * 16 + j * 4, ld4(er + j * 4));
    }
    WAIT_ASYNC();
    WAIT_DS();

    // GEMM1: [16x160] x [160x32] -> two 16x16 accumulators
    v8f acc0 = {0,0,0,0,0,0,0,0}, acc1 = {0,0,0,0,0,0,0,0};
    #pragma unroll
    for (int kt = 0; kt < 5; ++kt) {
        const int kb = kt * 32;
        v16bf a  = frag_a(At + e * 160, kb + half * 16, kb + half * 16 + 8);
        v16bf b0 = frag_b(wt1a + (e     ) * 160 + kb + half * 16);
        v16bf b1 = frag_b(wt1a + (e + 16) * 160 + kb + half * 16);
        acc0 = wmma_bf16(a, b0, acc0);
        acc1 = wmma_bf16(a, b1, acc1);
    }

    // bias + ReLU -> bf16 LDS h1 tile [16][32]
    unsigned short* Ht = sH + wv * (16 * 32);
    {
        const float bias0 = b1a[e], bias1 = b1a[e + 16];
        #pragma unroll
        for (int rr = 0; rr < 8; ++rr) {
            Ht[(half * 8 + rr) * 32 + e     ] = f2bf(fmaxf(acc0[rr] + bias0, 0.0f));
            Ht[(half * 8 + rr) * 32 + e + 16] = f2bf(fmaxf(acc1[rr] + bias1, 0.0f));
        }
    }
    WAIT_DS();

    // GEMM2: [16x32] x [32x32]
    v8f d0 = {0,0,0,0,0,0,0,0}, d1 = {0,0,0,0,0,0,0,0};
    {
        v16bf a  = frag_a(Ht + e * 32, half * 16, half * 16 + 8);
        v16bf b0 = frag_b(wt1b + (e     ) * 32 + half * 16);
        v16bf b1 = frag_b(wt1b + (e + 16) * 32 + half * 16);
        d0 = wmma_bf16(a, b0, d0);
        d1 = wmma_bf16(a, b1, d1);
    }

    // scatter-add into destination nodes
    const float bb0 = b1b[e], bb1 = b1b[e + 16];
    #pragma unroll
    for (int rr = 0; rr < 8; ++rr) {
        const int m = half * 8 + rr;
        if (tile * 16 + m < E) {
            const int node = sCol[wv * 16 + m];
            atomic_add_f32(&summed[(long)node * 32 + e     ], d0[rr] + bb0);
            atomic_add_f32(&summed[(long)node * 32 + e + 16], d1[rr] + bb1);
        }
    }
}

// ---------- kernel 4: node MLP ----------
// One 16-node tile per wave. 4 waves / block (LDS budget).

#define NODE_WAVES 4

__global__ __launch_bounds__(128) void node_kernel(
    const unsigned short* __restrict__ xb,           // [N][128] bf16
    const unsigned short* __restrict__ ub,           // [G][64]  bf16
    const int* __restrict__ batch,
    const float* __restrict__ b2a, const float* __restrict__ b2b,
    const unsigned short* __restrict__ wt2a, const unsigned short* __restrict__ wt2b,
    const float* __restrict__ summed, const float* __restrict__ cnt,
    float* __restrict__ out, int N, int nTiles)
{
    __shared__ __attribute__((aligned(16))) unsigned short sZ[NODE_WAVES * 16 * 224];
    __shared__ __attribute__((aligned(16))) unsigned short sH3[NODE_WAVES * 16 * 64];

    const int wv   = threadIdx.x >> 5;
    const int lane = threadIdx.x & 31;
    const int tile = blockIdx.x * NODE_WAVES + wv;
    if (tile >= nTiles) return;

    const int e    = lane & 15;
    const int half = lane >> 4;
    const int i    = tile * 16 + e;
    const int ii   = (i < N) ? i : 0;

    unsigned short* Zt = sZ + wv * (16 * 224);

    // x part: async bf16 row copy
    {
        const unsigned ldst = lds_addr(Zt + e * 224 + half * 64);
        const unsigned voff = (unsigned)ii * 256u + (unsigned)half * 128u;
        ASYNC_CP_HALFROW(ldst, voff, xb);
    }
    // agg part: summed/cnt, f32 -> bf16
    {
        const float rinv = 1.0f / fmaxf(cnt[ii], 1.0f);
        const float* sr = summed + (long)ii * 32 + half * 16;
        #pragma unroll
        for (int j = 0; j < 4; ++j) {
            float4 v = ld4(sr + j * 4);
            v.x *= rinv; v.y *= rinv; v.z *= rinv; v.w *= rinv;
            st4bf(Zt + e * 224 + 128 + half * 16 + j * 4, v);
        }
    }
    // u part: bf16 copy from pre-converted table
    {
        const unsigned short* ur = ub + (long)batch[ii] * 64 + half * 32;
        #pragma unroll
        for (int j = 0; j < 4; ++j)
            cp16(Zt + e * 224 + 160 + half * 32 + j * 8, ur + j * 8);
    }
    WAIT_ASYNC();
    WAIT_DS();

    // GEMM3: [16x224] x [224x64] -> 4 accumulators
    v8f acc[4];
    #pragma unroll
    for (int t = 0; t < 4; ++t) acc[t] = (v8f){0,0,0,0,0,0,0,0};
    #pragma unroll
    for (int kt = 0; kt < 7; ++kt) {
        const int kb = kt * 32;
        v16bf a = frag_a(Zt + e * 224, kb + half * 16, kb + half * 16 + 8);
        #pragma unroll
        for (int t = 0; t < 4; ++t) {
            v16bf b = frag_b(wt2a + (e + t * 16) * 224 + kb + half * 16);
            acc[t] = wmma_bf16(a, b, acc[t]);
        }
    }

    // bias + ReLU -> bf16 LDS h3 tile [16][64]
    unsigned short* Ht = sH3 + wv * (16 * 64);
    #pragma unroll
    for (int t = 0; t < 4; ++t) {
        const float bias = b2a[e + t * 16];
        #pragma unroll
        for (int rr = 0; rr < 8; ++rr)
            Ht[(half * 8 + rr) * 64 + t * 16 + e] = f2bf(fmaxf(acc[t][rr] + bias, 0.0f));
    }
    WAIT_DS();

    // GEMM4: [16x64] x [64x128] -> 8 accumulators
    v8f d[8];
    #pragma unroll
    for (int t = 0; t < 8; ++t) d[t] = (v8f){0,0,0,0,0,0,0,0};
    #pragma unroll
    for (int kt = 0; kt < 2; ++kt) {
        const int kb = kt * 32;
        v16bf a = frag_a(Ht + e * 64, kb + half * 16, kb + half * 16 + 8);
        #pragma unroll
        for (int t = 0; t < 8; ++t) {
            v16bf b = frag_b(wt2b + (e + t * 16) * 64 + kb + half * 16);
            d[t] = wmma_bf16(a, b, d[t]);
        }
    }

    // bias + store output [N][128]
    #pragma unroll
    for (int t = 0; t < 8; ++t) {
        const float bias = b2b[e + t * 16];
        #pragma unroll
        for (int rr = 0; rr < 8; ++rr) {
            const int m = half * 8 + rr;
            const int node = tile * 16 + m;
            if (node < N)
                out[(long)node * 128 + t * 16 + e] = d[t][rr] + bias;
        }
    }
}

// ---------- launcher ----------

extern "C" void kernel_launch(void* const* d_in, const int* in_sizes, int n_in,
                              void* d_out, int out_size, void* d_ws, size_t ws_size,
                              hipStream_t stream) {
    const float* x          = (const float*)d_in[0];
    const int*   edge_index = (const int*)  d_in[1];
    const float* edge_attr  = (const float*)d_in[2];
    const float* u          = (const float*)d_in[3];
    const int*   batch      = (const int*)  d_in[4];
    const float* W1a = (const float*)d_in[5];
    const float* b1a = (const float*)d_in[6];
    const float* W1b = (const float*)d_in[7];
    const float* b1b = (const float*)d_in[8];
    const float* W2a = (const float*)d_in[9];
    const float* b2a = (const float*)d_in[10];
    const float* W2b = (const float*)d_in[11];
    const float* b2b = (const float*)d_in[12];

    const int  N  = in_sizes[0] / 128;   // nodes
    const int  E  = in_sizes[1] / 2;     // edges
    const long Ux = in_sizes[3];         // global-feature element count (G*64)

    // workspace layout: bf16 weight tables, bf16 x/u tables, then scatter buffers
    unsigned short* wt1a = (unsigned short*)d_ws;           // [32][160]
    unsigned short* wt1b = wt1a + 32 * 160;                 // [32][32]
    unsigned short* wt2a = wt1b + 32 * 32;                  // [64][224]
    unsigned short* wt2b = wt2a + 64 * 224;                 // [128][64]
    unsigned short* xb   = wt2b + 128 * 64;                 // [N][128] bf16
    unsigned short* ub   = xb + (size_t)N * 128;            // [G][64]  bf16
    size_t off = (size_t)(28672 + (size_t)N * 128 + Ux);    // ushort units
    off = (off + 31) & ~(size_t)31;                         // 64B align
    float* summed = (float*)((unsigned short*)d_ws + off);  // [N][32]
    float* cnt    = summed + (size_t)N * 32;                // [N]

    // 0) weight conversion (28672 elements)
    prep_weights<<<(28672 + 255) / 256, 256, 0, stream>>>(
        W1a, W1b, W2a, W2b, wt1a, wt1b, wt2a, wt2b);

    // 1) x and u -> bf16 tables
    const long xq = (long)N * 32;        // float4 quads in x
    prep_cast<<<(unsigned)((xq + 255) / 256), 256, 0, stream>>>(x, xb, xq);
    const long uq = Ux / 4;
    prep_cast<<<(unsigned)((uq + 255) / 256), 256, 0, stream>>>(u, ub, uq);

    // 2) zero scatter buffers (N*33 floats)
    const int zn = N * 33;
    zero_scatter<<<(zn + 255) / 256, 256, 0, stream>>>(summed, zn);

    // 3) edge MLP + scatter
    const int eTiles = (E + 15) / 16;
    edge_kernel<<<(eTiles + EDGE_WAVES - 1) / EDGE_WAVES, 32 * EDGE_WAVES, 0, stream>>>(
        xb, edge_index, edge_attr, b1a, b1b, wt1a, wt1b, summed, cnt, E, eTiles);

    // 4) node MLP
    const int nTiles = (N + 15) / 16;
    node_kernel<<<(nTiles + NODE_WAVES - 1) / NODE_WAVES, 32 * NODE_WAVES, 0, stream>>>(
        xb, ub, batch, b2a, b2b, wt2a, wt2b, summed, cnt, (float*)d_out, N, nTiles);
}
